// MultiTaskPolicy_76055280877948
// MI455X (gfx1250) — matrix operand order repaired and checked by reading, not compiled, not run on hardware
//
#include <hip/hip_runtime.h>

typedef __attribute__((ext_vector_type(16))) _Float16 v16h;
typedef __attribute__((ext_vector_type(8)))  _Float16 v8h;
typedef __attribute__((ext_vector_type(8)))  float    v8f;
typedef __attribute__((ext_vector_type(4)))  int      v4i;

#define AS1 __attribute__((address_space(1)))
#define AS3 __attribute__((address_space(3)))

// Async global->LDS path (CDNA5). Guarded so host pass / older toolchains
// fall back to plain ld/st staging without breaking the compile.
#if defined(__AMDGCN__) && __has_builtin(__builtin_amdgcn_global_load_async_to_lds_b128) && __has_builtin(__builtin_amdgcn_s_wait_asynccnt)
#define USE_ASYNC_LDS 1
#else
#define USE_ASYNC_LDS 0
#endif

// ---------------- workspace layout (bytes) ----------------
static const size_t OFF_VPOOL = 0;                 // 256*256 f32
static const size_t OFF_PART  = 262144;            // 256*256*8 f32 (region 4MB)
static const size_t OFF_ACT1  = 4456448;           // 256*64*64*64 f16 NHWC  (134 MB)
static const size_t OFF_ACT2  = 138674176;         // 256*32*32*128 f16 NHWC (67 MB)
static const size_t OFF_B2    = 205783040;         // 128*576 f16  [N][K]
static const size_t OFF_B3    = 205930496;         // 256*1152 f16 [N][K]
static const size_t WS_NEED   = 206520320;

// ---------------- weight prep: OIHW f32 -> [N][K] f16, K = (ky,kx)*C + c ----
__global__ void __launch_bounds__(256)
prep_b2_kernel(const float* __restrict__ w, _Float16* __restrict__ B)
{
    int i = blockIdx.x * 256 + threadIdx.x;          // 128*576 = 73728
    if (i >= 128 * 576) return;
    int n = i / 576, k = i - n * 576;
    int pos = k >> 6, c = k & 63;                    // k = pos*64 + c
    B[i] = (_Float16)w[((size_t)n * 64 + c) * 9 + pos];
}

__global__ void __launch_bounds__(256)
prep_b3_kernel(const float* __restrict__ w, _Float16* __restrict__ B)
{
    int i = blockIdx.x * 256 + threadIdx.x;          // 256*1152 = 294912
    if (i >= 256 * 1152) return;
    int n = i / 1152, k = i - n * 1152;
    int pos = k >> 7, c = k & 127;                   // k = pos*128 + c
    B[i] = (_Float16)w[((size_t)n * 128 + c) * 9 + pos];
}

// ---------------- conv1: 3->64, stride 2, pad 1, naive VALU, f16 NHWC out ---
__global__ void __launch_bounds__(256)
conv1_kernel(const float* __restrict__ img, const float* __restrict__ w,
             const float* __restrict__ bias, _Float16* __restrict__ act1)
{
    const int idx = blockIdx.x * 256 + threadIdx.x;  // 256*64*64*4 threads
    const int cg = idx & 3;                          // 16-channel group
    const int ox = (idx >> 2) & 63;
    const int oy = (idx >> 8) & 63;
    const int n  = idx >> 14;
    float acc[16];
    #pragma unroll
    for (int j = 0; j < 16; ++j) acc[j] = bias[cg * 16 + j];
    for (int ci = 0; ci < 3; ++ci)
        for (int ky = 0; ky < 3; ++ky) {
            const int iy = oy * 2 - 1 + ky;
            if ((unsigned)iy >= 128u) continue;
            for (int kx = 0; kx < 3; ++kx) {
                const int ix = ox * 2 - 1 + kx;
                if ((unsigned)ix >= 128u) continue;
                const float pix = img[(((size_t)n * 3 + ci) * 128 + iy) * 128 + ix];
                const int wi = ci * 9 + ky * 3 + kx;
                #pragma unroll
                for (int j = 0; j < 16; ++j)
                    acc[j] += pix * w[(cg * 16 + j) * 27 + wi];
            }
        }
    v8h a0, a1;
    #pragma unroll
    for (int j = 0; j < 8; ++j) {
        float x0 = acc[j];     a0[j] = (_Float16)(x0 > 0.0f ? x0 : 0.0f);
        float x1 = acc[j + 8]; a1[j] = (_Float16)(x1 > 0.0f ? x1 : 0.0f);
    }
    _Float16* dst = act1 + ((((size_t)n * 64 + oy) * 64 + ox) << 6) + cg * 16;
    *(v8h*)dst = a0;
    *(v8h*)(dst + 8) = a1;
}

// ---------------- shared helpers for WMMA convs ----------------------------

// Stage a 32-row x KHALF-half [N][K] weight slice into LDS with padded rows
// (ROWH halfs) for bank-conflict-free reads. Async copy when available.
template<int KHALF, int ROWH>
__device__ __forceinline__
void stage_B_tile(const _Float16* __restrict__ g, _Float16* sB, int tid)
{
    constexpr int CPR = KHALF / 8;        // 16-byte chunks per row
    constexpr int TOT = 32 * CPR;
#if USE_ASYNC_LDS
    const unsigned lbase = (unsigned)(uintptr_t)(void*)sB;
    #pragma unroll 1
    for (int idx = tid; idx < TOT; idx += 256) {
        const int row = idx / CPR;
        const int col = idx - row * CPR;
        AS1 v4i* gp = (AS1 v4i*)(uintptr_t)(const void*)
                      (g + (size_t)row * KHALF + col * 8);
        AS3 v4i* lp = (AS3 v4i*)(lbase + (unsigned)(row * ROWH + col * 8) * 2u);
        __builtin_amdgcn_global_load_async_to_lds_b128(gp, lp, 0, 0);
    }
    __builtin_amdgcn_s_wait_asynccnt(0);
#else
    #pragma unroll 1
    for (int idx = tid; idx < TOT; idx += 256) {
        const int row = idx / CPR;
        const int col = idx - row * CPR;
        *(v8h*)(sB + (size_t)row * ROWH + col * 8) =
            *(const v8h*)(g + (size_t)row * KHALF + col * 8);
    }
#endif
    __syncthreads();
}

// A fragment: 16(M)x32(K) f16 from NHWC activations. base already includes
// pixel offset + c0 + kh*8; zero when the patch position is out of bounds.
__device__ __forceinline__ v16h load_a_frag(const _Float16* base, bool inb)
{
    v16h a;
    if (inb) {
        const v8h lo = *(const v8h*)(base);        // K = c0+8kh .. +7
        const v8h hi = *(const v8h*)(base + 16);   // K = c0+16+8kh .. +7
        #pragma unroll
        for (int i = 0; i < 8; ++i) { a[i] = lo[i]; a[i + 8] = hi[i]; }
    } else {
        #pragma unroll
        for (int i = 0; i < 16; ++i) a[i] = (_Float16)0.0f;
    }
    return a;
}

__device__ __forceinline__ v16h load_b_frag(const _Float16* bp)
{
    const v8h b0 = *(const v8h*)(bp);              // K = kb+8kh .. +7
    const v8h b1 = *(const v8h*)(bp + 16);         // K = kb+16+8kh .. +7
    v16h b;
    #pragma unroll
    for (int i = 0; i < 8; ++i) { b[i] = b0[i]; b[i + 8] = b1[i]; }
    return b;
}

// ---------------- conv2: 64->128, 32x32 out, implicit GEMM via WMMA --------
// WG tile: 128(M) x 32(N); 8 waves as 4(M) x 2(N); each wave: 2 M-subtiles.
__global__ void __launch_bounds__(256)
conv2_wmma(const _Float16* __restrict__ act1, const _Float16* __restrict__ B2,
           const float* __restrict__ bias, _Float16* __restrict__ act2)
{
    constexpr int ROWH = 584;                        // 1168B padded row (292 dw)
    __shared__ __attribute__((aligned(32))) _Float16 sB[32 * ROWH];

    const int lane = threadIdx.x & 31;
    const int wave = threadIdx.x >> 5;
    const int l15  = lane & 15;
    const int kh   = lane >> 4;                      // K-half selector

    const int bid  = blockIdx.x;                     // 256 imgs * 32 WGs
    const int n    = bid >> 5;
    const int r    = bid & 31;
    const int mblk = r & 7;                          // 8 M-blocks of 128 pixels
    const int nblk = r >> 3;                         // 4 N-blocks of 32 channels
    const int wm   = wave & 3, wn = wave >> 2;
    const int mbase = mblk * 128 + wm * 32;          // wave covers 32 pixels
    const int ch    = nblk * 32 + wn * 16 + l15;     // out channel

    stage_B_tile<576, ROWH>(B2 + (size_t)(nblk * 32) * 576, sB, threadIdx.x);

    const int m0 = mbase + l15,      m1 = mbase + 16 + l15;
    const int oy0 = m0 >> 5, ox0 = m0 & 31;
    const int oy1 = m1 >> 5, ox1 = m1 & 31;

    const _Float16* brow = sB + (size_t)(wn * 16 + l15) * ROWH + kh * 8;

    v8f acc0 = {}, acc1 = {};
    for (int pos = 0; pos < 9; ++pos) {
        const int ky = pos / 3, kx = pos % 3;
        const int iy0 = oy0 * 2 - 1 + ky, ix0 = ox0 * 2 - 1 + kx;
        const int iy1 = oy1 * 2 - 1 + ky, ix1 = ox1 * 2 - 1 + kx;
        const bool inb0 = ((unsigned)iy0 < 64u) & ((unsigned)ix0 < 64u);
        const bool inb1 = ((unsigned)iy1 < 64u) & ((unsigned)ix1 < 64u);
        #pragma unroll
        for (int chf = 0; chf < 2; ++chf) {
            const int c0 = chf * 32;
            const int kb = pos * 64 + c0;
            const v16h a0 = load_a_frag(
                act1 + ((((size_t)n * 64 + iy0) * 64 + ix0) << 6) + c0 + kh * 8, inb0);
            const v16h a1 = load_a_frag(
                act1 + ((((size_t)n * 64 + iy1) * 64 + ix1) << 6) + c0 + kh * 8, inb1);
            const v16h b = load_b_frag(brow + kb);   // shared for both M-subtiles
            acc0 = __builtin_amdgcn_wmma_f32_16x16x32_f16(
                false, a0, false, b, (short)0, acc0, false, false);
            acc1 = __builtin_amdgcn_wmma_f32_16x16x32_f16(
                false, a1, false, b, (short)0, acc1, false, false);
        }
    }
    const float bv = bias[ch];
    #pragma unroll
    for (int t = 0; t < 2; ++t) {
        const v8f acc = t ? acc1 : acc0;
        #pragma unroll
        for (int j = 0; j < 8; ++j) {
            const int mm = mbase + t * 16 + j + kh * 8;   // C/D row map
            const int zy = mm >> 5, zx = mm & 31;
            float x = acc[j] + bv;
            x = x > 0.0f ? x : 0.0f;
            act2[((((size_t)n * 32 + zy) * 32 + zx) << 7) + ch] = (_Float16)x;
        }
    }
}

// ---------------- conv3: 128->256, 16x16 out, WMMA + fused avg-pool --------
// WG tile: 128(M) x 32(N); 8 waves as 4(M) x 2(N); each wave: 2 M-subtiles.
__global__ void __launch_bounds__(256)
conv3_wmma(const _Float16* __restrict__ act2, const _Float16* __restrict__ B3,
           const float* __restrict__ bias, float* __restrict__ part)
{
    constexpr int ROWH = 1160;                       // 2320B padded row (580 dw)
    __shared__ __attribute__((aligned(32))) _Float16 sB[32 * ROWH];

    const int lane = threadIdx.x & 31;
    const int wave = threadIdx.x >> 5;
    const int l15  = lane & 15;
    const int kh   = lane >> 4;

    const int bid  = blockIdx.x;                     // 256 imgs * 16 WGs
    const int n    = bid >> 4;
    const int r    = bid & 15;
    const int mblk = r & 1;                          // 2 M-blocks of 128 pixels
    const int nblk = r >> 1;                         // 8 N-blocks of 32 channels
    const int wm   = wave & 3, wn = wave >> 2;
    const int mbase = mblk * 128 + wm * 32;
    const int ch    = nblk * 32 + wn * 16 + l15;

    stage_B_tile<1152, ROWH>(B3 + (size_t)(nblk * 32) * 1152, sB, threadIdx.x);

    const int m0 = mbase + l15,      m1 = mbase + 16 + l15;
    const int oy0 = m0 >> 4, ox0 = m0 & 15;
    const int oy1 = m1 >> 4, ox1 = m1 & 15;

    const _Float16* brow = sB + (size_t)(wn * 16 + l15) * ROWH + kh * 8;

    v8f acc0 = {}, acc1 = {};
    for (int pos = 0; pos < 9; ++pos) {
        const int ky = pos / 3, kx = pos % 3;
        const int iy0 = oy0 * 2 - 1 + ky, ix0 = ox0 * 2 - 1 + kx;
        const int iy1 = oy1 * 2 - 1 + ky, ix1 = ox1 * 2 - 1 + kx;
        const bool inb0 = ((unsigned)iy0 < 32u) & ((unsigned)ix0 < 32u);
        const bool inb1 = ((unsigned)iy1 < 32u) & ((unsigned)ix1 < 32u);
        #pragma unroll
        for (int chf = 0; chf < 4; ++chf) {
            const int c0 = chf * 32;
            const int kb = pos * 128 + c0;
            const v16h a0 = load_a_frag(
                act2 + ((((size_t)n * 32 + iy0) * 32 + ix0) << 7) + c0 + kh * 8, inb0);
            const v16h a1 = load_a_frag(
                act2 + ((((size_t)n * 32 + iy1) * 32 + ix1) << 7) + c0 + kh * 8, inb1);
            const v16h b = load_b_frag(brow + kb);
            acc0 = __builtin_amdgcn_wmma_f32_16x16x32_f16(
                false, a0, false, b, (short)0, acc0, false, false);
            acc1 = __builtin_amdgcn_wmma_f32_16x16x32_f16(
                false, a1, false, b, (short)0, acc1, false, false);
        }
    }
    // bias + ReLU per pixel, then sum this wave's 32 pixels per channel
    const float bv = bias[ch];
    float s = 0.0f;
    #pragma unroll
    for (int j = 0; j < 8; ++j) {
        float x0 = acc0[j] + bv; s += x0 > 0.0f ? x0 : 0.0f;
        float x1 = acc1[j] + bv; s += x1 > 0.0f ? x1 : 0.0f;
    }
    s += __shfl_xor(s, 16, 32);                      // combine the two K-halves' rows
    if (lane < 16) {
        const int mt = mblk * 4 + wm;                // 8 pixel-tiles of 32 per image
        part[((size_t)n * 256 + ch) * 8 + mt] = s;   // ordered partials (deterministic)
    }
}

__global__ void __launch_bounds__(256)
pool_reduce_kernel(const float* __restrict__ part, float* __restrict__ vpool)
{
    int i = blockIdx.x * 256 + threadIdx.x;          // 256*256
    float s = 0.0f;
    #pragma unroll
    for (int t = 0; t < 8; ++t) s += part[(size_t)i * 8 + t];
    vpool[i] = s * (1.0f / 256.0f);
}

// ---------------- fused proprio MLP + fusion + routed task heads -----------
__global__ void __launch_bounds__(256)
heads_kernel(const float* __restrict__ vpool, const float* __restrict__ prop,
             const int* __restrict__ tids,
             const float* __restrict__ p1w, const float* __restrict__ p1b,
             const float* __restrict__ p2w, const float* __restrict__ p2b,
             const float* __restrict__ f1w, const float* __restrict__ f1b,
             const float* __restrict__ f2w, const float* __restrict__ f2b,
             const float* __restrict__ temb,
             const float* __restrict__ gw,  const float* __restrict__ gb,
             const float* __restrict__ h1w, const float* __restrict__ h1b,
             const float* __restrict__ h2w, const float* __restrict__ h2b,
             const float* __restrict__ h3w, const float* __restrict__ h3b,
             float* __restrict__ out)
{
    __shared__ float s_a[320];
    __shared__ float s_b[256];
    __shared__ float s_c[8];
    const int n = blockIdx.x;
    const int o = threadIdx.x;
    const int task = tids[n];

    if (o < 256) s_a[o] = vpool[n * 256 + o];
    if (o < 7)   s_c[o] = prop[n * 7 + o];
    __syncthreads();
    if (o < 64) {                                    // p1 + ReLU
        float acc = p1b[o];
        for (int i = 0; i < 7; ++i) acc += s_c[i] * p1w[i * 64 + o];
        s_b[o] = fmaxf(acc, 0.0f);
    }
    __syncthreads();
    if (o < 64) {                                    // p2 (no ReLU)
        float acc = p2b[o];
        for (int i = 0; i < 64; ++i) acc += s_b[i] * p2w[i * 64 + o];
        s_a[256 + o] = acc;
    }
    __syncthreads();
    {                                                // f1: [320]->256 + ReLU
        float acc = f1b[o];
        for (int i = 0; i < 320; ++i) acc += s_a[i] * f1w[i * 256 + o];
        s_b[o] = fmaxf(acc, 0.0f);
    }
    __syncthreads();
    {                                                // f2: 256->256 + ReLU
        float acc = f2b[o];
        for (int i = 0; i < 256; ++i) acc += s_b[i] * f2w[i * 256 + o];
        s_a[o] = fmaxf(acc, 0.0f);
    }
    if (o < 32) s_a[256 + o] = temb[task * 32 + o];  // task embedding
    __syncthreads();
    {                                                // g: [288]->256 + ReLU
        float acc = gb[o];
        for (int i = 0; i < 288; ++i) acc += s_a[i] * gw[i * 256 + o];
        s_b[o] = fmaxf(acc, 0.0f);
    }
    __syncthreads();
    if (o < 128) {                                   // h1 (routed) + ReLU
        float acc = h1b[task * 128 + o];
        const float* w = h1w + (size_t)task * 256 * 128;
        for (int i = 0; i < 256; ++i) acc += s_b[i] * w[i * 128 + o];
        s_a[o] = fmaxf(acc, 0.0f);
    }
    __syncthreads();
    if (o < 128) {                                   // h2 (routed) + ReLU
        float acc = h2b[task * 128 + o];
        const float* w = h2w + (size_t)task * 128 * 128;
        for (int i = 0; i < 128; ++i) acc += s_a[i] * w[i * 128 + o];
        s_b[o] = fmaxf(acc, 0.0f);
    }
    __syncthreads();
    if (o < 4) {                                     // h3 (routed), no ReLU
        float acc = h3b[task * 4 + o];
        const float* w = h3w + (size_t)task * 128 * 4;
        for (int i = 0; i < 128; ++i) acc += s_b[i] * w[i * 4 + o];
        out[n * 4 + o] = acc;
    }
}

extern "C" void kernel_launch(void* const* d_in, const int* in_sizes, int n_in,
                              void* d_out, int out_size, void* d_ws, size_t ws_size,
                              hipStream_t stream)
{
    (void)in_sizes; (void)n_in; (void)out_size;
    if (ws_size < WS_NEED) return;                   // need ~197 MiB scratch

    const float* images = (const float*)d_in[0];
    const float* prop   = (const float*)d_in[1];
    const int*   tids   = (const int*)d_in[2];
    const float* c1w = (const float*)d_in[3];  const float* c1b = (const float*)d_in[4];
    const float* c2w = (const float*)d_in[5];  const float* c2b = (const float*)d_in[6];
    const float* c3w = (const float*)d_in[7];  const float* c3b = (const float*)d_in[8];
    const float* p1w = (const float*)d_in[9];  const float* p1b = (const float*)d_in[10];
    const float* p2w = (const float*)d_in[11]; const float* p2b = (const float*)d_in[12];
    const float* f1w = (const float*)d_in[13]; const float* f1b = (const float*)d_in[14];
    const float* f2w = (const float*)d_in[15]; const float* f2b = (const float*)d_in[16];
    const float* temb = (const float*)d_in[17];
    const float* gw  = (const float*)d_in[18]; const float* gb  = (const float*)d_in[19];
    const float* h1w = (const float*)d_in[20]; const float* h1b = (const float*)d_in[21];
    const float* h2w = (const float*)d_in[22]; const float* h2b = (const float*)d_in[23];
    const float* h3w = (const float*)d_in[24]; const float* h3b = (const float*)d_in[25];

    char* ws = (char*)d_ws;
    float*    vpool = (float*)   (ws + OFF_VPOOL);
    float*    part  = (float*)   (ws + OFF_PART);
    _Float16* act1  = (_Float16*)(ws + OFF_ACT1);
    _Float16* act2  = (_Float16*)(ws + OFF_ACT2);
    _Float16* B2    = (_Float16*)(ws + OFF_B2);
    _Float16* B3    = (_Float16*)(ws + OFF_B3);

    prep_b2_kernel<<<288, 256, 0, stream>>>(c2w, B2);
    prep_b3_kernel<<<1152, 256, 0, stream>>>(c3w, B3);
    conv1_kernel<<<16384, 256, 0, stream>>>(images, c1w, c1b, act1);
    conv2_wmma<<<8192, 256, 0, stream>>>(act1, B2, c2b, act2);
    conv3_wmma<<<4096, 256, 0, stream>>>(act2, B3, c3b, part);
    pool_reduce_kernel<<<256, 256, 0, stream>>>(part, vpool);
    heads_kernel<<<256, 256, 0, stream>>>(vpool, prop, tids,
        p1w, p1b, p2w, p2b, f1w, f1b, f2w, f2b, temb, gw, gb,
        h1w, h1b, h2w, h2b, h3w, h3b, (float*)d_out);
}
